// LocalFWL_71528385348008
// MI455X (gfx1250) — compile-verified, matrix operand order checked by reading, not run on hardware
//
#include <hip/hip_runtime.h>
#include <hip/hip_bf16.h>
#include <stdint.h>

// Problem constants (match reference)
static constexpr int GN = 2048;   // nodes
static constexpr int GF = 128;    // input features
static constexpr int GH = 32;     // hidden
static constexpr int GE = 65536;  // edges
static constexpr int GP = 768;    // query pairs

typedef __attribute__((ext_vector_type(16))) _Float16 v16h;
typedef __attribute__((ext_vector_type(8)))  _Float16 v8h;
typedef __attribute__((ext_vector_type(8)))  float    v8f;

// ---------------------------------------------------------------------------
// WMMA fragment load (CDNA5 wave32, 16-bit operand, 16x16x32 shape).
// Works for both A (row-major 16xK, idx=row) and B stored K-transposed in LDS
// (sBt[n][k], idx=col): lane's 16 halves are two contiguous 16B runs:
//   lane L<16 : elems 0..7 = K(k0+0..7),  8..15 = K(k0+16..23)
//   lane L>=16: elems 0..7 = K(k0+8..15), 8..15 = K(k0+24..31)
// => two ds_load_b128 per lane.
// ---------------------------------------------------------------------------
__device__ __forceinline__ v16h load_frag16(const _Float16* base, int stride,
                                            int idx, int k0, int lane) {
    const int kb = (lane < 16) ? 0 : 8;
    const _Float16* p = base + idx * stride + k0 + kb;
    v8h lo = *(const v8h*)p;
    v8h hi = *(const v8h*)(p + 16);
    return __builtin_shufflevector(lo, hi, 0, 1, 2, 3, 4, 5, 6, 7,
                                   8, 9, 10, 11, 12, 13, 14, 15);
}

// ---------------------------------------------------------------------------
// Init kernels
// ---------------------------------------------------------------------------
__global__ void k_init_emap(int* __restrict__ emap, int total) {
    int i = blockIdx.x * 256 + threadIdx.x;
    if (i < total) emap[i] = -1;
}

__global__ void k_init_h(float* __restrict__ h1, const float* __restrict__ b1,
                         float* __restrict__ h2, const float* __restrict__ b2,
                         float* __restrict__ deg) {
    int i = blockIdx.x * 256 + threadIdx.x;
    if (i < GN * GH) {
        h1[i] = b1[i & (GH - 1)];
        h2[i] = b2[i & (GH - 1)];
    }
    if (i < GN) deg[i] = 0.0f;
}

__global__ void k_deg(const int* __restrict__ dst, float* __restrict__ deg) {
    int e = blockIdx.x * 256 + threadIdx.x;
    if (e < GE) atomicAdd(&deg[dst[e]], 1.0f);
}

__global__ void k_dinv(float* __restrict__ deg) {
    int i = blockIdx.x * 256 + threadIdx.x;
    if (i < GN) deg[i] = rsqrtf(deg[i] + 1.0f);  // +1 self-loop; always > 0
}

__global__ void k_emap_fill(int* __restrict__ emap, const int* __restrict__ src,
                            const int* __restrict__ dst) {
    int e = blockIdx.x * 256 + threadIdx.x;
    if (e < GE) emap[src[e] * GN + dst[e]] = e;  // duplicate (s,d): last writer wins
}

// ---------------------------------------------------------------------------
// Dense GEMM (M x K) @ (K x 32) -> (M x 32), one wave per 16-row tile, WMMA.
// B is staged transposed (sBt[n][k]) so fragment loads are ds_load_b128 pairs.
// ---------------------------------------------------------------------------
template <int K>
__global__ __launch_bounds__(32) void k_mm_wmma(const float* __restrict__ A,
                                                const float* __restrict__ B,
                                                float* __restrict__ C) {
    __shared__ alignas(16) _Float16 sA[16 * K];
    __shared__ alignas(16) _Float16 sBt[32 * K];
    const int lane = threadIdx.x;
    const int m0   = blockIdx.x * 16;

    // stage A rows (K-contiguous)
    for (int i = lane; i < 16 * K; i += 32)
        sA[i] = (_Float16)A[(m0 + (i / K)) * K + (i % K)];
    // stage B transposed: lane owns column n=lane, walks K contiguously
    for (int i = lane; i < K * 32; i += 32) {
        int k = i >> 5, n = i & 31;
        sBt[n * K + k] = (_Float16)B[i];
    }
    __syncthreads();

#pragma unroll
    for (int nt = 0; nt < 2; ++nt) {
        v8f c = {};
#pragma unroll
        for (int k0 = 0; k0 < K; k0 += 32) {
            v16h a = load_frag16(sA, K, lane & 15, k0, lane);
            v16h b = load_frag16(sBt, K, nt * 16 + (lane & 15), k0, lane);
            c = __builtin_amdgcn_wmma_f32_16x16x32_f16(false, a, false, b,
                                                       (short)0, c, false, false);
        }
        int col   = nt * 16 + (lane & 15);
        int rbase = (lane < 16) ? 0 : 8;
#pragma unroll
        for (int r = 0; r < 8; ++r)
            C[(m0 + rbase + r) * 32 + col] = c[r];
    }
}

// ---------------------------------------------------------------------------
// Normalized scatter aggregation: h[d] += t[s]*dinv[s]*dinv[d] (+ self loops)
// ---------------------------------------------------------------------------
__global__ void k_agg(const float* __restrict__ t, const int* __restrict__ src,
                      const int* __restrict__ dst, const float* __restrict__ dinv,
                      float* __restrict__ h) {
    int idx = blockIdx.x * 256 + threadIdx.x;
    if (idx < GE * GH) {
        int e = idx >> 5, hh = idx & 31;
        int s = src[e], d = dst[e];
        atomicAdd(&h[d * GH + hh], t[s * GH + hh] * dinv[s] * dinv[d]);
    } else {
        idx -= GE * GH;
        if (idx < GN * GH) {
            int n = idx >> 5, hh = idx & 31;
            float di = dinv[n];
            atomicAdd(&h[n * GH + hh], t[n * GH + hh] * di * di);
        }
    }
}

// ---------------------------------------------------------------------------
// Edge MLP: xe = h2[s]*h2[d]; both branches y = LN(xe@W + b)*g + beta, relu.
// One wave per 16 edges; WMMA with transposed-W staging; LN on LDS spill.
// ---------------------------------------------------------------------------
__global__ __launch_bounds__(32) void k_edge_mlp(
    const float* __restrict__ h2, const int* __restrict__ src, const int* __restrict__ dst,
    const float* __restrict__ w1, const float* __restrict__ bb1,
    const float* __restrict__ g1, const float* __restrict__ be1,
    const float* __restrict__ w2, const float* __restrict__ bb2,
    const float* __restrict__ g2, const float* __restrict__ be2,
    float* __restrict__ x1, float* __restrict__ x2) {
    __shared__ alignas(16) _Float16 sA[16 * 32];
    __shared__ alignas(16) _Float16 sWt[32 * 32];  // transposed: sWt[n][k]
    __shared__ alignas(16) float    sY[16 * 32];

    const int lane = threadIdx.x;
    const int e0   = blockIdx.x * 16;
    const int er   = lane & 15;
    const int fh   = (lane >> 4) * 16;
    const int e    = e0 + er;
    const int s    = src[e], d = dst[e];

#pragma unroll
    for (int f = 0; f < 16; ++f) {
        float v = h2[s * GH + fh + f] * h2[d * GH + fh + f];
        sA[er * 32 + fh + f] = (_Float16)v;
    }
    __syncthreads();

    const v16h a = load_frag16(sA, 32, lane & 15, 0, lane);

    for (int br = 0; br < 2; ++br) {
        const float* W  = br ? w2 : w1;
        const float* B  = br ? bb2 : bb1;
        const float* G  = br ? g2 : g1;
        const float* Be = br ? be2 : be1;
        float*       X  = br ? x2 : x1;

        // stage W transposed: lane owns column n=lane
#pragma unroll
        for (int k = 0; k < 32; ++k) sWt[lane * 32 + k] = (_Float16)W[k * 32 + lane];
        __syncthreads();

#pragma unroll
        for (int nt = 0; nt < 2; ++nt) {
            float bias = B[nt * 16 + (lane & 15)];
            v8f c;
#pragma unroll
            for (int r = 0; r < 8; ++r) c[r] = bias;
            v16h b = load_frag16(sWt, 32, nt * 16 + (lane & 15), 0, lane);
            c = __builtin_amdgcn_wmma_f32_16x16x32_f16(false, a, false, b,
                                                       (short)0, c, false, false);
            int col   = nt * 16 + (lane & 15);
            int rbase = (lane < 16) ? 0 : 8;
#pragma unroll
            for (int r = 0; r < 8; ++r)
                sY[(rbase + r) * 32 + col] = c[r];
        }
        __syncthreads();

        if (lane < 16) {  // LayerNorm + relu over the 32 outputs of row `lane`
            float y[32], mu = 0.0f;
#pragma unroll
            for (int j = 0; j < 32; ++j) { y[j] = sY[lane * 32 + j]; mu += y[j]; }
            mu *= (1.0f / 32.0f);
            float var = 0.0f;
#pragma unroll
            for (int j = 0; j < 32; ++j) { float t = y[j] - mu; var += t * t; }
            var *= (1.0f / 32.0f);
            float inv = rsqrtf(var + 1e-5f);
#pragma unroll
            for (int j = 0; j < 32; ++j) {
                float o = (y[j] - mu) * inv * G[j] + Be[j];
                X[(e0 + lane) * GH + j] = fmaxf(o, 0.0f);
            }
        }
        __syncthreads();
    }
}

// ---------------------------------------------------------------------------
// xx[p] = h2[pos0[p]] * h2[pos1[p]]
// ---------------------------------------------------------------------------
__global__ void k_xx(const float* __restrict__ h2, const int* __restrict__ pos,
                     float* __restrict__ xx) {
    int i = blockIdx.x * 256 + threadIdx.x;
    if (i < GP * GH) {
        int p = i >> 5, hh = i & 31;
        xx[i] = h2[pos[p] * GH + hh] * h2[pos[GP + p] * GH + hh];
    }
}

// ---------------------------------------------------------------------------
// pos_val[p,h] = sum_n x2[e(u,n),h] * x1[e(n,v),h] via dense edge-id map.
// ---------------------------------------------------------------------------
__global__ __launch_bounds__(256) void k_posval(const int* __restrict__ emap,
                                                const int* __restrict__ pos,
                                                const float* __restrict__ x1,
                                                const float* __restrict__ x2,
                                                float* __restrict__ pv) {
    __shared__ float acc[GH];
    const int p = blockIdx.x;
    const int u = pos[p];
    const int v = pos[GP + p];
    if (threadIdx.x < GH) acc[threadIdx.x] = 0.0f;
    __syncthreads();
    for (int n = threadIdx.x; n < GN; n += 256) {
        int e1 = emap[u * GN + n];
        if (e1 >= 0) {
            int e2 = emap[n * GN + v];
            if (e2 >= 0) {
#pragma unroll
                for (int hh = 0; hh < GH; ++hh)
                    atomicAdd(&acc[hh], x2[e1 * GH + hh] * x1[e2 * GH + hh]);
            }
        }
    }
    __syncthreads();
    if (threadIdx.x < GH) pv[p * GH + threadIdx.x] = acc[threadIdx.x];
}

// ---------------------------------------------------------------------------
// Final MLP: z=[pv,xx] (64) -> relu(z@W1+b1) (32) -> @W2+b2 -> out[p]
// ---------------------------------------------------------------------------
__global__ __launch_bounds__(32) void k_final(const float* __restrict__ pv,
                                              const float* __restrict__ xx,
                                              const float* __restrict__ w1,
                                              const float* __restrict__ b1,
                                              const float* __restrict__ w2,
                                              const float* __restrict__ b2,
                                              float* __restrict__ out) {
    const int p = blockIdx.x, h = threadIdx.x;
    float s = b1[h];
#pragma unroll 4
    for (int j = 0; j < GH; ++j) s += pv[p * GH + j] * w1[j * GH + h];
#pragma unroll 4
    for (int j = 0; j < GH; ++j) s += xx[p * GH + j] * w1[(GH + j) * GH + h];
    s = fmaxf(s, 0.0f);
    float t = s * w2[h];
#pragma unroll
    for (int m = 16; m; m >>= 1) t += __shfl_xor(t, m, 32);
    if (h == 0) out[p] = t + b2[0];
}

// ---------------------------------------------------------------------------
extern "C" void kernel_launch(void* const* d_in, const int* in_sizes, int n_in,
                              void* d_out, int out_size, void* d_ws, size_t ws_size,
                              hipStream_t stream) {
    (void)in_sizes; (void)n_in; (void)out_size; (void)ws_size;

    const float* x      = (const float*)d_in[0];
    const int*   ei     = (const int*)d_in[1];
    const int*   pos    = (const int*)d_in[2];
    const float* W1     = (const float*)d_in[3];
    const float* b1     = (const float*)d_in[4];
    const float* W2     = (const float*)d_in[5];
    const float* b2     = (const float*)d_in[6];
    const float* m1_w   = (const float*)d_in[7];
    const float* m1_b   = (const float*)d_in[8];
    const float* m1_g   = (const float*)d_in[9];
    const float* m1_be  = (const float*)d_in[10];
    const float* m2_w   = (const float*)d_in[11];
    const float* m2_b   = (const float*)d_in[12];
    const float* m2_g   = (const float*)d_in[13];
    const float* m2_be  = (const float*)d_in[14];
    const float* m3_w1  = (const float*)d_in[15];
    const float* m3_b1  = (const float*)d_in[16];
    const float* m3_w2  = (const float*)d_in[17];
    const float* m3_b2  = (const float*)d_in[18];
    float*       out    = (float*)d_out;

    const int* src = ei;
    const int* dst = ei + GE;

    // workspace carve-up (~34 MB total)
    uintptr_t base = (uintptr_t)d_ws;
    auto alloc = [&](size_t bytes) {
        void* p = (void*)base;
        base += (bytes + 255) & ~(size_t)255;
        return p;
    };
    float* dinv = (float*)alloc((size_t)GN * 4);
    float* t1   = (float*)alloc((size_t)GN * GH * 4);
    float* h1   = (float*)alloc((size_t)GN * GH * 4);
    float* t2   = (float*)alloc((size_t)GN * GH * 4);
    float* h2   = (float*)alloc((size_t)GN * GH * 4);
    float* x1   = (float*)alloc((size_t)GE * GH * 4);
    float* x2   = (float*)alloc((size_t)GE * GH * 4);
    float* xx   = (float*)alloc((size_t)GP * GH * 4);
    float* pv   = (float*)alloc((size_t)GP * GH * 4);
    int*   emap = (int*)  alloc((size_t)GN * GN * 4);

    // 1. init
    k_init_emap<<<(GN * GN + 255) / 256, 256, 0, stream>>>(emap, GN * GN);
    k_init_h<<<(GN * GH + 255) / 256, 256, 0, stream>>>(h1, b1, h2, b2, dinv);
    // 2. degrees -> dinv
    k_deg<<<(GE + 255) / 256, 256, 0, stream>>>(dst, dinv);
    k_dinv<<<(GN + 255) / 256, 256, 0, stream>>>(dinv);
    // 3. conv1: t1 = x @ W1 (K=128), aggregate into h1 (pre-seeded with b1)
    k_mm_wmma<GF><<<GN / 16, 32, 0, stream>>>(x, W1, t1);
    k_agg<<<((GE + GN) * GH + 255) / 256, 256, 0, stream>>>(t1, src, dst, dinv, h1);
    // 4. conv2: t2 = h1 @ W2 (K=32), aggregate into h2 (pre-seeded with b2)
    k_mm_wmma<GH><<<GN / 16, 32, 0, stream>>>(h1, W2, t2);
    k_agg<<<((GE + GN) * GH + 255) / 256, 256, 0, stream>>>(t2, src, dst, dinv, h2);
    // 5. edge-id map for sparse einsum
    k_emap_fill<<<(GE + 255) / 256, 256, 0, stream>>>(emap, src, dst);
    // 6. per-edge MLPs (WMMA) -> x1, x2
    k_edge_mlp<<<GE / 16, 32, 0, stream>>>(h2, src, dst,
                                           m1_w, m1_b, m1_g, m1_be,
                                           m2_w, m2_b, m2_g, m2_be,
                                           x1, x2);
    // 7. query features
    k_xx<<<(GP * GH + 255) / 256, 256, 0, stream>>>(h2, pos, xx);
    k_posval<<<GP, 256, 0, stream>>>(emap, pos, x1, x2, pv);
    // 8. final MLP -> out[768]
    k_final<<<GP, 32, 0, stream>>>(pv, xx, m3_w1, m3_b1, m3_w2, m3_b2, out);
}
